// Int4Einsum_40913858462382
// MI455X (gfx1250) — matrix operand order repaired and checked by reading, compiled, not run on hardware
//
#include <hip/hip_runtime.h>
#include <stdint.h>

typedef _Float16 v16h __attribute__((ext_vector_type(16)));
typedef float    v8f  __attribute__((ext_vector_type(8)));
typedef unsigned int u32x4 __attribute__((ext_vector_type(4)));
typedef int          i32x8 __attribute__((ext_vector_type(8)));
typedef int          i32x4 __attribute__((ext_vector_type(4)));

// Problem constants (fixed by the reference)
constexpr int D_DIM = 4096;
constexpr int F_DIM = 8192;
constexpr int K_EXP = 2;
constexpr int M_DIM = 4 * 2048;    // B*T
constexpr int G_DIM = 32;          // D / GROUP_SIZE

// Tiling
constexpr int BM = 128, BN = 128, BK = 32;
constexpr int NIT = D_DIM / BK;    // 128 K-steps

struct BStage { int4 q[4]; float4 s; float4 zs; };

// One TDM descriptor-driven DMA: x tile [BM rows x BK f32], row stride D_DIM, -> LDS.
// D# per CDNA5 ISA ch.8: group0 = {count/type/lds/global}, group1 = {dims/tile/strides}.
__device__ __forceinline__ void tdm_load_x_tile(unsigned ldsAddr, const float* gptr) {
    const uint64_t ga = (uint64_t)(uintptr_t)gptr;
    u32x4 g0;
    g0[0] = 1u;                                   // count=1 (valid user descriptor)
    g0[1] = ldsAddr;                              // lds_addr (bytes)
    g0[2] = (unsigned)ga;                         // global_addr[31:0]
    g0[3] = (unsigned)((ga >> 32) & 0x01FFFFFFu)  // global_addr[56:32]
          | (2u << 30);                           // type=2 ("image")
    i32x8 g1;
    g1[0] = (int)(2u << 16);                      // workgroup_mask=0 (not in cluster), data_size=2 -> 4B
    g1[1] = (int)((unsigned)BK << 16);            // tensor_dim0[15:0] at bits 63:48  (=32)
    g1[2] = (int)((unsigned)BM << 16);            // tensor_dim0[31:16]=0 | tensor_dim1[15:0]=128
    g1[3] = (int)((unsigned)BK << 16);            // tensor_dim1[31:16]=0 | tile_dim0=32
    g1[4] = (int)(unsigned)BM;                    // tile_dim1=128, tile_dim2=0
    g1[5] = (int)(unsigned)D_DIM;                 // tensor_dim0_stride[31:0] = 4096 elements
    g1[6] = 0;                                    // stride0[47:32]=0 | tensor_dim1_stride lo=0
    g1[7] = 0;
    const i32x4 z4 = {0, 0, 0, 0};                // groups 2/3 unused (2-D tile)
    const i32x8 z8 = {0, 0, 0, 0, 0, 0, 0, 0};
    __builtin_amdgcn_tensor_load_to_lds(g0, g1, z4, z4, z8, 0);
}

__global__ __launch_bounds__(256)
void int4_gemm_wmma(const float* __restrict__ x,
                    const int*   __restrict__ qw,
                    const int*   __restrict__ qz,
                    const float* __restrict__ sc,
                    float* __restrict__ out)
{
    // Ping-pong LDS: A raw f32 (filled by TDM), B dequantized f16 transposed [n][k]
    __shared__ __align__(16) float    As[2][BM * BK];   // 2 x 16 KB
    __shared__ __align__(32) _Float16 Bs[2][BN * BK];   // 2 x  8 KB

    const int tid  = threadIdx.x;
    const int lane = tid & 31;
    const int wid  = tid >> 5;
    const int wm   = wid & 3;      // wave's 32-row slab (0..3)
    const int wn   = wid >> 2;     // wave's 64-col slab (0..1)

    const int m0 = blockIdx.x * BM;   // fastest-varying: weight tile stays hot in L2
    const int n0 = blockIdx.y * BN;
    const int ke = blockIdx.z;

    // B staging map: 32 f-chunks of 4 columns, 8 d-rows per pass
    const int bfc  = (tid & 31) * 4;   // f_local 0,4,...,124
    const int brow = tid >> 5;         // 0..7

    const int mlo = lane & 15;
    const int khi = lane >> 4;         // 0 or 1

    const unsigned aBase[2] = { (unsigned)(uintptr_t)&As[0][0],
                                (unsigned)(uintptr_t)&As[1][0] };

    auto load_b = [&](int d0, BStage& r) {
        const int g = d0 >> 7;  // quant group; constant within one K-step
        r.s = *(const float4*)&sc[((size_t)ke * G_DIM + g) * F_DIM + n0 + bfc];
        const int4 z = *(const int4*)&qz[((size_t)ke * G_DIM + g) * F_DIM + n0 + bfc];
        r.zs.x = -(float)z.x * r.s.x;   // fold (q-z)*s = fma(q, s, -z*s)
        r.zs.y = -(float)z.y * r.s.y;
        r.zs.z = -(float)z.z * r.s.z;
        r.zs.w = -(float)z.w * r.s.w;
#pragma unroll
        for (int p = 0; p < 4; ++p) {
            const int dl = p * 8 + brow;  // 0..31
            r.q[p] = *(const int4*)&qw[((size_t)ke * D_DIM + d0 + dl) * F_DIM + n0 + bfc];
        }
    };

    auto commit_b = [&](int buf, const BStage& r) {
        _Float16* b = &Bs[buf][0];
#pragma unroll
        for (int p = 0; p < 4; ++p) {
            const int dl = p * 8 + brow;
            b[(bfc + 0) * BK + dl] = (_Float16)__builtin_fmaf((float)r.q[p].x, r.s.x, r.zs.x);
            b[(bfc + 1) * BK + dl] = (_Float16)__builtin_fmaf((float)r.q[p].y, r.s.y, r.zs.y);
            b[(bfc + 2) * BK + dl] = (_Float16)__builtin_fmaf((float)r.q[p].z, r.s.z, r.zs.z);
            b[(bfc + 3) * BK + dl] = (_Float16)__builtin_fmaf((float)r.q[p].w, r.s.w, r.zs.w);
        }
    };

    // A fragment: contiguous K order (k = khi*16 + e), matching B; convert f32 LDS -> f16 at load time
    auto load_a_frag = [&](int buf, int i) -> v16h {
        const int row = wm * 32 + i * 16 + mlo;
        const float* p = &As[buf][row * BK];
        v8f c0 = *(const v8f*)(p + khi * 16);       // K khi*16 .. +7
        v8f c1 = *(const v8f*)(p + khi * 16 + 8);   // K khi*16+8 .. +7
        v16h a;
#pragma unroll
        for (int e = 0; e < 8; ++e) {
            a[e]     = (_Float16)c0[e];
            a[8 + e] = (_Float16)c1[e];
        }
        return a;
    };

    v8f acc[2][4] = {};
    BStage br;

    // ---------- prologue: stage tile 0 ----------
    if (wid == 0) tdm_load_x_tile(aBase[0], x + (size_t)m0 * D_DIM);
    load_b(0, br);
    commit_b(0, br);
    if (wid == 0) __builtin_amdgcn_s_wait_tensorcnt(0);
    __syncthreads();

    // ---------- main loop: double-buffered, TDM runs under WMMA ----------
    for (int it = 0; it < NIT; ++it) {
        const int cur   = it & 1;
        const bool more = (it + 1) < NIT;

        if (more) {
            if (wid == 0)
                tdm_load_x_tile(aBase[cur ^ 1], x + (size_t)m0 * D_DIM + (it + 1) * BK);
            load_b((it + 1) * BK, br);
        }

        // ---- consume current tile ----
        v16h bfrag[4];
#pragma unroll
        for (int j = 0; j < 4; ++j) {
            const int col = wn * 64 + j * 16 + mlo;
            bfrag[j] = *(const v16h*)&Bs[cur][col * BK + khi * 16];
        }
#pragma unroll
        for (int i = 0; i < 2; ++i) {
            const v16h a = load_a_frag(cur, i);
#pragma unroll
            for (int j = 0; j < 4; ++j) {
                acc[i][j] = __builtin_amdgcn_wmma_f32_16x16x32_f16(
                    false, a, false, bfrag[j], (short)0, acc[i][j], false, false);
            }
        }

        if (more) commit_b(cur ^ 1, br);
        if (more && wid == 0) __builtin_amdgcn_s_wait_tensorcnt(0);
        __syncthreads();
    }

    // ---------- epilogue: C/D layout -> out[(m*K + ke)*F + f] ----------
#pragma unroll
    for (int i = 0; i < 2; ++i) {
#pragma unroll
        for (int j = 0; j < 4; ++j) {
            const int f = n0 + wn * 64 + j * 16 + mlo;
#pragma unroll
            for (int r = 0; r < 8; ++r) {
                const int m = m0 + wm * 32 + i * 16 + r + khi * 8;
                out[((size_t)m * K_EXP + ke) * F_DIM + f] = acc[i][j][r];
            }
        }
    }
}

extern "C" void kernel_launch(void* const* d_in, const int* in_sizes, int n_in,
                              void* d_out, int out_size, void* d_ws, size_t ws_size,
                              hipStream_t stream) {
    const float* x  = (const float*)d_in[0];
    const int*   qw = (const int*)d_in[1];
    const int*   qz = (const int*)d_in[2];
    const float* sc = (const float*)d_in[3];
    float* out = (float*)d_out;

    dim3 grid(M_DIM / BM, F_DIM / BN, K_EXP);  // m fastest: weight tile reused out of L2
    int4_gemm_wmma<<<grid, dim3(256, 1, 1), 0, stream>>>(x, qw, qz, sc, out);
}
